// Classifier_24008867184845
// MI455X (gfx1250) — compile-verified, matrix operand order checked by reading, not run on hardware
//
#include <hip/hip_runtime.h>
#include <hip/hip_bf16.h>
#include <cstdint>

typedef _Float16 half4  __attribute__((ext_vector_type(4)));
typedef _Float16 half8  __attribute__((ext_vector_type(8)));
typedef _Float16 half16 __attribute__((ext_vector_type(16)));
typedef float    f32x8  __attribute__((ext_vector_type(8)));

union Frag16 { half16 v; half8 h[2]; };

// ---------------------------------------------------------------------------
// CDNA5 async global->LDS helpers (ASYNCcnt-tracked, bypass VGPRs).
// LDS destination address = low 32 bits of the generic pointer (flat LDS
// aperture rule).
// ---------------------------------------------------------------------------
__device__ __forceinline__ void async_load_b128(uint32_t lds_off, const void* gaddr)
{
    asm volatile("global_load_async_to_lds_b128 %0, %1, off"
                 :: "v"(lds_off), "v"((uint64_t)(uintptr_t)gaddr)
                 : "memory");
}
__device__ __forceinline__ void wait_async0() { asm volatile("s_wait_asynccnt 0x0" ::: "memory"); }
__device__ __forceinline__ void wait_async3() { asm volatile("s_wait_asynccnt 0x3" ::: "memory"); }

// ---------------------------------------------------------------------------
// Kernel 1: per-channel 2D integral image, stored channel-innermost:
// integ[(rr*65+cc)*512 + ch], rr/cc in [0,64], border row/col = 0.
// ---------------------------------------------------------------------------
__global__ __launch_bounds__(64) void integral_kernel(const float* __restrict__ base,
                                                      float* __restrict__ integ)
{
    __shared__ float tile[64][65];
    const int ch  = blockIdx.x;
    const int tid = threadIdx.x;
    const float* src = base + (size_t)ch * 4096;
    for (int idx = tid; idx < 4096; idx += 64)
        tile[idx >> 6][idx & 63] = src[idx];
    __syncthreads();
    { float s = 0.f; for (int j = 0; j < 64; ++j) { s += tile[tid][j]; tile[tid][j] = s; } }
    __syncthreads();
    { float s = 0.f; for (int i = 0; i < 64; ++i) { s += tile[i][tid]; tile[i][tid] = s; } }
    __syncthreads();
    for (int idx = tid; idx < 65 * 65; idx += 64) {
        int rr = idx / 65, cc = idx % 65;
        float v = (rr > 0 && cc > 0) ? tile[rr - 1][cc - 1] : 0.f;
        integ[(size_t)idx * 512 + ch] = v;
    }
}

// ---------------------------------------------------------------------------
// f32 -> f16 bulk convert (vectorized by 4)
// ---------------------------------------------------------------------------
__global__ __launch_bounds__(256) void cvt_f32_f16_kernel(const float* __restrict__ src,
                                                          _Float16* __restrict__ dst, int n4)
{
    int i = blockIdx.x * 256 + threadIdx.x;
    if (i >= n4) return;
    float4 q = ((const float4*)src)[i];
    half4 h = { (_Float16)q.x, (_Float16)q.y, (_Float16)q.z, (_Float16)q.w };
    ((half4*)dst)[i] = h;
}

// ---------------------------------------------------------------------------
// Pack [d3_w(21,512); d4_w(80,512); zeros] -> wcat[128][512] (f16)
// ---------------------------------------------------------------------------
__global__ __launch_bounds__(256) void packcat_kernel(const float* __restrict__ d3,
                                                      const float* __restrict__ d4,
                                                      _Float16* __restrict__ wcat)
{
    int idx = blockIdx.x * 256 + threadIdx.x;       // 128*512 = 65536 total
    if (idx >= 128 * 512) return;
    int n = idx >> 9, k = idx & 511;
    float v = 0.f;
    if (n < 21)       v = d3[n * 512 + k];
    else if (n < 101) v = d4[(n - 21) * 512 + k];
    wcat[idx] = (_Float16)v;
}

// ---------------------------------------------------------------------------
// Kernel 2: fused ROI adaptive-avg-pool (via integral image) + channel
// reduction GEMM: flat[r][o*49 + bin] = pooled[r][:,bin] . conv_w[o,:] + b
// One block per ROI. M=49(pad 64) x N=128 x K=512 WMMA, channel-chunked.
// conv weights pre-converted to f16 (cwh) and staged via async-to-LDS.
// ---------------------------------------------------------------------------
__global__ __launch_bounds__(256) void roiconv_kernel(const float* __restrict__ integ,
                                                      const int*   __restrict__ rois,
                                                      const _Float16* __restrict__ cwh,
                                                      const float* __restrict__ conv_b,
                                                      _Float16*    __restrict__ flat)
{
    __shared__ _Float16 sPool[64][264];   // 49 bins (pad 64) x 256-ch chunk (pad 8)
    __shared__ _Float16 sB[128][40];      // 128 out-ch x 32 K (pad 8)
    __shared__ int   sOff[4][49];
    __shared__ float sInv[49];

    const int r    = blockIdx.x;
    const int tid  = threadIdx.x;
    const int lane = tid & 31;
    const int wave = tid >> 5;
    const int wm   = wave & 3;      // M tile (16 rows each)
    const int wn   = wave >> 2;     // N half (64 cols each)

    if (tid < 49) {
        int bi = tid / 7, bj = tid % 7;
        int x = rois[r*4+0], y = rois[r*4+1], hh = rois[r*4+2], ww = rois[r*4+3];
        int r0 = x + (bi * ww) / 7;
        int r1 = x + ((bi + 1) * ww + 6) / 7;     // ceil
        int c0 = y + (bj * hh) / 7;
        int c1 = y + ((bj + 1) * hh + 6) / 7;
        sOff[0][tid] = (r0 * 65 + c0) * 512;
        sOff[1][tid] = (r0 * 65 + c1) * 512;
        sOff[2][tid] = (r1 * 65 + c0) * 512;
        sOff[3][tid] = (r1 * 65 + c1) * 512;
        sInv[tid] = 1.0f / (float)((r1 - r0) * (c1 - c0));
    }

    // async staging addresses for the conv-weight tile (per-thread 32B)
    const int brow = tid >> 1;
    const int bcg  = (tid & 1) * 16;
    const uint32_t lB = (uint32_t)(uintptr_t)&sB[brow][bcg];
    const _Float16* bbase = cwh + (size_t)brow * 512 + bcg;

    f32x8 acc[4];
    const f32x8 zero8 = {0.f,0.f,0.f,0.f,0.f,0.f,0.f,0.f};
    for (int j = 0; j < 4; ++j) acc[j] = zero8;

    for (int chunk = 0; chunk < 2; ++chunk) {
        const int chbase = chunk * 256;
        __syncthreads();
        // pooled values for this 256-channel chunk; thread = channel
        for (int b = 0; b < 64; ++b) {
            _Float16 hv = (_Float16)0.f;
            if (b < 49) {
                int c = chbase + tid;
                float v = (integ[sOff[3][b] + c] - integ[sOff[1][b] + c]
                         - integ[sOff[2][b] + c] + integ[sOff[0][b] + c]) * sInv[b];
                hv = (_Float16)v;
            }
            sPool[b][tid] = hv;
        }
        __syncthreads();

        for (int kb = 0; kb < 8; ++kb) {
            // stage conv-weight tile [128 x 32] f16 asynchronously
            const _Float16* bs = bbase + chbase + kb * 32;
            async_load_b128(lB,      bs);
            async_load_b128(lB + 16, bs + 8);
            wait_async0();
            __syncthreads();

            const int k0 = (lane & 16) ? 8 : 0;
            Frag16 fa, fb[4];
            fa.h[0] = *(const half8*)&sPool[wm*16 + (lane & 15)][kb*32 + k0];
            fa.h[1] = *(const half8*)&sPool[wm*16 + (lane & 15)][kb*32 + k0 + 16];
            for (int j = 0; j < 4; ++j) {
                int nrow = wn*64 + j*16 + (lane & 15);
                fb[j].h[0] = *(const half8*)&sB[nrow][k0];
                fb[j].h[1] = *(const half8*)&sB[nrow][k0 + 16];
            }
            for (int j = 0; j < 4; ++j)
                acc[j] = __builtin_amdgcn_wmma_f32_16x16x32_f16(false, fa.v, false, fb[j].v,
                                                                (short)0, acc[j], false, false);
            __syncthreads();
        }
    }

    const int mhi = (lane & 16) ? 8 : 0;
    for (int j = 0; j < 4; ++j) {
        int o = wn*64 + j*16 + (lane & 15);
        float bv = conv_b[o];
        _Float16* dst = flat + (size_t)r * 6272 + o * 49;
        for (int v = 0; v < 8; ++v) {
            int m = wm*16 + mhi + v;
            if (m < 49) dst[m] = (_Float16)(acc[j][v] + bv);
        }
    }
}

// ---------------------------------------------------------------------------
// Generic GEMM: Out[M,N] = act(A[M,K](f16) * B[N,K](f16)^T + bias)
// Block tile 128x64, 8 waves (32x32 each, 2x2 wmma frags), K step 32.
// Double-buffered LDS; tiles staged with async global->LDS (prefetch of the
// next K-tile overlaps the current tile's WMMAs).
// All dims are multiples of the tile sizes for this workload.
// ---------------------------------------------------------------------------
template<bool RELU, bool OUT_HALF>
__global__ __launch_bounds__(256) void gemm_kernel(const _Float16* __restrict__ A,
                                                   const _Float16* __restrict__ B,
                                                   const float*    __restrict__ bias,
                                                   void*           __restrict__ Out,
                                                   int M, int N, int K)
{
    __shared__ _Float16 sA[2][128][40];
    __shared__ _Float16 sB[2][64][40];

    const int tid  = threadIdx.x;
    const int lane = tid & 31;
    const int wave = tid >> 5;
    const int wm   = wave & 3;
    const int wn   = wave >> 2;
    const int m0   = blockIdx.y * 128;
    const int n0   = blockIdx.x * 64;

    f32x8 acc[2][2];
    const f32x8 zero8 = {0.f,0.f,0.f,0.f,0.f,0.f,0.f,0.f};
    for (int i = 0; i < 2; ++i)
        for (int j = 0; j < 2; ++j) acc[i][j] = zero8;

    const int arow = tid >> 1;          // 0..127
    const int acg  = (tid & 1) * 16;    // 16 halves (2 x b128)
    const int brow = tid >> 2;          // 0..63
    const int bcg  = (tid & 3) * 8;     // 8 halves (1 x b128)

    const _Float16* aptr = A + (size_t)(m0 + arow) * K + acg;
    const _Float16* bptr = B + (size_t)(n0 + brow) * K + bcg;
    const uint32_t lA = (uint32_t)(uintptr_t)&sA[0][arow][acg];
    const uint32_t lB = (uint32_t)(uintptr_t)&sB[0][brow][bcg];
    constexpr uint32_t A_BUF = 128 * 40 * 2;   // bytes per A buffer
    constexpr uint32_t B_BUF = 64 * 40 * 2;    // bytes per B buffer

    const int KB = K >> 5;
    // prologue: stage K-tile 0 into buffer 0
    async_load_b128(lA,      aptr);
    async_load_b128(lA + 16, aptr + 8);
    async_load_b128(lB,      bptr);

    for (int kb = 0; kb < KB; ++kb) {
        const int p = kb & 1;
        if (kb + 1 < KB) {
            // prefetch next K-tile into the other buffer
            const uint32_t ao = ((kb + 1) & 1) ? A_BUF : 0u;
            const uint32_t bo = ((kb + 1) & 1) ? B_BUF : 0u;
            const _Float16* an = aptr + (size_t)(kb + 1) * 32;
            const _Float16* bn = bptr + (size_t)(kb + 1) * 32;
            async_load_b128(lA + ao,      an);
            async_load_b128(lA + ao + 16, an + 8);
            async_load_b128(lB + bo,      bn);
            wait_async3();               // current tile (oldest 3) complete
        } else {
            wait_async0();
        }
        __syncthreads();

        const int k0 = (lane & 16) ? 8 : 0;
        Frag16 fa[2], fb[2];
        for (int i = 0; i < 2; ++i) {
            int mr = wm*32 + i*16 + (lane & 15);
            fa[i].h[0] = *(const half8*)&sA[p][mr][k0];
            fa[i].h[1] = *(const half8*)&sA[p][mr][k0 + 16];
        }
        for (int j = 0; j < 2; ++j) {
            int nr = wn*32 + j*16 + (lane & 15);
            fb[j].h[0] = *(const half8*)&sB[p][nr][k0];
            fb[j].h[1] = *(const half8*)&sB[p][nr][k0 + 16];
        }
        for (int i = 0; i < 2; ++i)
            for (int j = 0; j < 2; ++j)
                acc[i][j] = __builtin_amdgcn_wmma_f32_16x16x32_f16(false, fa[i].v, false, fb[j].v,
                                                                   (short)0, acc[i][j], false, false);
        __syncthreads();   // closes WAR hazard vs. next iteration's async writes
    }

    const int mhi = (lane & 16) ? 8 : 0;
    for (int i = 0; i < 2; ++i) {
        for (int j = 0; j < 2; ++j) {
            int gn = n0 + wn*32 + j*16 + (lane & 15);
            float bv = bias ? bias[gn] : 0.f;
            for (int v = 0; v < 8; ++v) {
                int gm = m0 + wm*32 + i*16 + mhi + v;
                float val = acc[i][j][v] + bv;
                if (RELU) val = fmaxf(val, 0.f);
                if (OUT_HALF) ((_Float16*)Out)[(size_t)gm * N + gn] = (_Float16)val;
                else          ((float*)Out)[(size_t)gm * N + gn] = val;
            }
        }
    }
}

// ---------------------------------------------------------------------------
// Final: softmax over first 21 logits -> out_class; copy next 80 -> out_regr.
// d_out = [2048*21 class | 2048*80 regr] (f32).
// ---------------------------------------------------------------------------
__global__ __launch_bounds__(128) void softmax_out_kernel(const float* __restrict__ logits,
                                                          float* __restrict__ out)
{
    __shared__ float sl[21];
    __shared__ float sred[2];
    const int r = blockIdx.x, tid = threadIdx.x;
    const float* lg = logits + (size_t)r * 128;
    if (tid < 21) sl[tid] = lg[tid];
    if (tid < 80) out[2048 * 21 + (size_t)r * 80 + tid] = lg[21 + tid];
    __syncthreads();
    if (tid == 0) {
        float mx = sl[0];
        for (int j = 1; j < 21; ++j) mx = fmaxf(mx, sl[j]);
        float s = 0.f;
        for (int j = 0; j < 21; ++j) s += __expf(sl[j] - mx);
        sred[0] = mx; sred[1] = s;
    }
    __syncthreads();
    if (tid < 21) out[(size_t)r * 21 + tid] = __expf(sl[tid] - sred[0]) / sred[1];
}

// ---------------------------------------------------------------------------
extern "C" void kernel_launch(void* const* d_in, const int* in_sizes, int n_in,
                              void* d_out, int out_size, void* d_ws, size_t ws_size,
                              hipStream_t stream)
{
    const float* base_x = (const float*)d_in[0];
    const int*   rois   = (const int*)  d_in[1];
    const float* conv_w = (const float*)d_in[2];
    const float* conv_b = (const float*)d_in[3];
    const float* d1_w   = (const float*)d_in[4];
    const float* d1_b   = (const float*)d_in[5];
    const float* d2_w   = (const float*)d_in[6];
    const float* d2_b   = (const float*)d_in[7];
    const float* d3_w   = (const float*)d_in[8];
    const float* d4_w   = (const float*)d_in[9];
    float* out = (float*)d_out;

    // workspace layout (all 256B aligned)
    constexpr size_t OFF_INTEG  = 0;                          // 65*65*512*4   =  8,652,800
    constexpr size_t OFF_FLAT   = OFF_INTEG  +  8652800;      // 2048*6272*2   = 25,690,112
    constexpr size_t OFF_H1     = OFF_FLAT   + 25690112;      // 2048*1024*2   =  4,194,304
    constexpr size_t OFF_H2     = OFF_H1     +  4194304;      // 2048*512*2    =  2,097,152
    constexpr size_t OFF_W1H    = OFF_H2     +  2097152;      // 1024*6272*2   = 12,845,056
    constexpr size_t OFF_W2H    = OFF_W1H    + 12845056;      // 512*1024*2    =  1,048,576
    constexpr size_t OFF_CWH    = OFF_W2H    +  1048576;      // 128*512*2     =    131,072
    constexpr size_t OFF_WCATH  = OFF_CWH    +   131072;      // 128*512*2     =    131,072
    constexpr size_t OFF_LOGITS = OFF_WCATH  +   131072;      // 2048*128*4    =  1,048,576

    char* ws = (char*)d_ws;
    float*    integ  = (float*)   (ws + OFF_INTEG);
    _Float16* flat   = (_Float16*)(ws + OFF_FLAT);
    _Float16* h1     = (_Float16*)(ws + OFF_H1);
    _Float16* h2     = (_Float16*)(ws + OFF_H2);
    _Float16* w1h    = (_Float16*)(ws + OFF_W1H);
    _Float16* w2h    = (_Float16*)(ws + OFF_W2H);
    _Float16* cwh    = (_Float16*)(ws + OFF_CWH);
    _Float16* wcath  = (_Float16*)(ws + OFF_WCATH);
    float*    logits = (float*)   (ws + OFF_LOGITS);

    integral_kernel<<<512, 64, 0, stream>>>(base_x, integ);
    cvt_f32_f16_kernel<<<6272, 256, 0, stream>>>(d1_w, w1h, 1024 * 6272 / 4);
    cvt_f32_f16_kernel<<<512,  256, 0, stream>>>(d2_w, w2h,  512 * 1024 / 4);
    cvt_f32_f16_kernel<<<64,   256, 0, stream>>>(conv_w, cwh, 128 * 512 / 4);
    packcat_kernel<<<256, 256, 0, stream>>>(d3_w, d4_w, wcath);
    roiconv_kernel<<<2048, 256, 0, stream>>>(integ, rois, cwh, conv_b, flat);
    // h1 = relu(flat @ d1_w^T + d1_b)   M=2048 N=1024 K=6272
    gemm_kernel<true,  true ><<<dim3(16, 16), 256, 0, stream>>>(flat, w1h, d1_b, (void*)h1, 2048, 1024, 6272);
    // h2 = relu(h1 @ d2_w^T + d2_b)     M=2048 N=512  K=1024
    gemm_kernel<true,  true ><<<dim3(8, 16),  256, 0, stream>>>(h1,   w2h, d2_b, (void*)h2, 2048, 512, 1024);
    // logits = h2 @ [d3;d4;0]^T         M=2048 N=128  K=512
    gemm_kernel<false, false><<<dim3(2, 16),  256, 0, stream>>>(h2,   wcath, nullptr, (void*)logits, 2048, 128, 512);
    softmax_out_kernel<<<2048, 128, 0, stream>>>(logits, out);
}
